// GRU_8821862826732
// MI455X (gfx1250) — compile-verified, matrix operand order checked by reading, not run on hardware
//
#include <hip/hip_runtime.h>
#include <hip/hip_bf16.h>

typedef __bf16 bf16_t;
typedef __bf16 v16bf __attribute__((ext_vector_type(16)));
typedef float  v8f   __attribute__((ext_vector_type(8)));

#define GRU_B  32
#define GRU_S  2048
#define GRU_IN 256
#define GRU_H  512
#define SCAN_BLOCKS 16
#define SCAN_THREADS 384

// ---------------- helpers ----------------

__device__ inline bf16_t to_bf16(float f) {
    unsigned u = __builtin_bit_cast(unsigned, f);
    unsigned r = (u + 0x7fffu + ((u >> 16) & 1u)) >> 16;
    unsigned short s = (unsigned short)r;
    return __builtin_bit_cast(bf16_t, s);
}

// Async copy of 16 bytes global -> LDS (per-lane addresses), ASYNCcnt-tracked.
__device__ inline void async_b128_to_lds(void* lds_ptr, const void* gptr) {
    unsigned lds = (unsigned)(uintptr_t)lds_ptr;                 // low 32 bits = LDS offset
    unsigned long long ga = (unsigned long long)(uintptr_t)gptr;
    asm volatile("global_load_async_to_lds_b128 %0, %1, off"
                 :: "v"(lds), "v"(ga) : "memory");
}
__device__ inline void wait_asynccnt0() {
    asm volatile("s_wait_asynccnt 0x0" ::: "memory");
}

// Load one 16x32 bf16 WMMA operand fragment from a row-major [rows][ld] matrix
// (global or LDS; addrspace inferred after inlining -> ds_load_b128 for LDS).
// ISA 16-bit layout: lanes 0-15 hold K chunks {k0..k0+7, k0+16..k0+23};
// lanes 16-31 hold K chunks {k0+8..k0+15, k0+24..k0+31}.
__device__ inline v16bf load_frag(const bf16_t* __restrict__ base, int ld,
                                  int row0, int k0, int lane) {
    const int r    = lane & 15;
    const int half = lane >> 4;
    const bf16_t* p = base + (size_t)(row0 + r) * (size_t)ld + k0 + half * 8;
    union { uint4 u[2]; v16bf v; } u;
    u.u[0] = *(const uint4*)(p);
    u.u[1] = *(const uint4*)(p + 16);
    return u.v;
}

__device__ inline v8f wmma_bf16(v16bf a, v16bf b, v8f c) {
    return __builtin_amdgcn_wmma_f32_16x16x32_bf16(
        /*neg_a=*/false, a, /*neg_b=*/false, b,
        /*c_mod=*/(short)0, c, /*reuse_a=*/false, /*reuse_b=*/false);
}

__device__ inline float fast_sigmoid(float x) {
    return 1.0f / (1.0f + __expf(-x));
}

// Device-scope software barrier across all persistent blocks.
__device__ inline void grid_barrier(unsigned* ctr, unsigned expected) {
    __syncthreads();
    if (threadIdx.x == 0) {
        __threadfence();  // make rh/h stores visible at device scope
        __hip_atomic_fetch_add(ctr, 1u, __ATOMIC_ACQ_REL, __HIP_MEMORY_SCOPE_AGENT);
        unsigned v;
        do {
            v = __hip_atomic_load(ctr, __ATOMIC_ACQUIRE, __HIP_MEMORY_SCOPE_AGENT);
            if (v < expected) __builtin_amdgcn_s_sleep(2);
        } while (v < expected);
    }
    __syncthreads();
}

// ---------------- prep kernels ----------------

__global__ void f32_to_bf16_kernel(const float* __restrict__ src,
                                   bf16_t* __restrict__ dst, int n) {
    int i = blockIdx.x * blockDim.x + threadIdx.x;
    if (i < n) dst[i] = to_bf16(src[i]);
}

// src: f32 [K][N] row-major -> dst: bf16 [N][K] (transposed, K contiguous)
__global__ void wconv_kernel(const float* __restrict__ src,
                             bf16_t* __restrict__ dst, int K, int N) {
    int i = blockIdx.x * blockDim.x + threadIdx.x;
    if (i < K * N) {
        int k = i / N, n = i % N;
        dst[(size_t)n * K + k] = to_bf16(src[i]);
    }
}

__global__ void hinit_kernel(const float* __restrict__ hsrc,
                             bf16_t* __restrict__ h_bf,
                             float* __restrict__ h_f32,
                             unsigned* __restrict__ ctr) {
    int i = blockIdx.x * blockDim.x + threadIdx.x;
    if (i == 0) *ctr = 0u;
    if (i < GRU_B * GRU_H) {
        float v = hsrc[i];
        h_f32[i] = v;
        h_bf[i]  = to_bf16(v);
    }
}

// ---------------- input-projection GEMM (async-LDS staged, bf16 WMMA) -------
// out[m][n] = sum_k A[m][k] * Wt[n][k] + bias[n]
// Block: 256 thr = 8 waves (2m x 4n), 64x128 block tile, k-step 32,
// double-buffered GLOBAL_LOAD_ASYNC_TO_LDS_B128 staging, ds_load fragments.
__global__ __launch_bounds__(256) void gate_gemm_kernel(
    const bf16_t* __restrict__ A,   // [M][K]
    const bf16_t* __restrict__ Wt,  // [512][K]
    const float*  __restrict__ bias,// [512]
    float* __restrict__ out,        // [M][512]
    int M, int K) {
    const int wave = threadIdx.x >> 5;
    const int lane = threadIdx.x & 31;
    const int mblk = blockIdx.x * 64;
    const int nblk = blockIdx.y * 128;
    const int mo = (wave >> 2) * 32;   // wave M offset within block tile
    const int no = (wave & 3)  * 32;   // wave N offset within block tile
    (void)M;

    // padded: row stride 40 elems = 80B, 16B-aligned chunks
    __shared__ __align__(16) bf16_t sA[2][64][40];
    __shared__ __align__(16) bf16_t sB[2][128][40];

    // per-thread fixed staging coordinates (straight-line, no divergence)
    const int rowA = threadIdx.x >> 2, kcA = (threadIdx.x & 3) * 8;
    const int rowB0 = rowA;            // B rows 0..63
    const int rowB1 = rowA + 64;       // B rows 64..127
    const bf16_t* gA  = A  + (size_t)(mblk + rowA)  * K + kcA;
    const bf16_t* gB0 = Wt + (size_t)(nblk + rowB0) * K + kcA;
    const bf16_t* gB1 = Wt + (size_t)(nblk + rowB1) * K + kcA;

    auto stage = [&](int p, int k0) {
        async_b128_to_lds(&sA[p][rowA][kcA],  gA  + k0);
        async_b128_to_lds(&sB[p][rowB0][kcA], gB0 + k0);
        async_b128_to_lds(&sB[p][rowB1][kcA], gB1 + k0);
    };

    v8f acc[2][2] = {};
    stage(0, 0);
    wait_asynccnt0();
    __syncthreads();

    int p = 0;
    for (int k0 = 0; k0 < K; k0 += 32) {
        if (k0 + 32 < K) stage(p ^ 1, k0 + 32);
        v16bf a0 = load_frag(&sA[p][0][0], 40, mo,      0, lane);
        v16bf a1 = load_frag(&sA[p][0][0], 40, mo + 16, 0, lane);
        v16bf b0 = load_frag(&sB[p][0][0], 40, no,      0, lane);
        v16bf b1 = load_frag(&sB[p][0][0], 40, no + 16, 0, lane);
        acc[0][0] = wmma_bf16(a0, b0, acc[0][0]);
        acc[1][0] = wmma_bf16(a1, b0, acc[1][0]);
        acc[0][1] = wmma_bf16(a0, b1, acc[0][1]);
        acc[1][1] = wmma_bf16(a1, b1, acc[1][1]);
        wait_asynccnt0();
        __syncthreads();
        p ^= 1;
    }

    const int rlo = lane & 15, half = lane >> 4;
#pragma unroll
    for (int mi = 0; mi < 2; ++mi)
#pragma unroll
        for (int ni = 0; ni < 2; ++ni) {
            int n = nblk + no + ni * 16 + rlo;
            float bv = bias[n];
#pragma unroll
            for (int rr = 0; rr < 8; ++rr) {
                int m = mblk + mo + mi * 16 + rr + half * 8;
                out[(size_t)m * GRU_H + n] = acc[mi][ni][rr] + bv;
            }
        }
}

// ---------------- persistent recurrent scan ----------------
// 16 blocks x 384 thr (12 waves). Block b owns output columns [32b, 32b+32).
// Per phase, h (or r*h) [32x512] bf16 is async-copied once into LDS; all
// fragment reads are ds_load_b128 (stride 520 elems -> bank-conflict-free).
// Recurrent weight B-fragments live in VGPRs for the whole sequence.
__global__ __launch_bounds__(SCAN_THREADS, 1) void gru_scan_kernel(
    const float* __restrict__ gates,   // [3][B][S][H] precomputed x-projections+bias
    const bf16_t* __restrict__ Wzh,    // [H][H] (n-major, k contiguous)
    const bf16_t* __restrict__ Wrh,
    const bf16_t* __restrict__ Whh,
    bf16_t* __restrict__ h_bf,         // [B][H]
    float*  __restrict__ h_f32,        // [B][H]
    bf16_t* __restrict__ rh_bf,        // [B][H]
    bf16_t* __restrict__ y_bf,         // [B][S][H] or null
    float*  __restrict__ y_f32,        // [B][S][H] or null
    float*  __restrict__ hn_out,       // [B][H]
    unsigned* __restrict__ ctr) {
    const int wave  = threadIdx.x >> 5;
    const int lane  = threadIdx.x & 31;
    const int nbase = blockIdx.x * 32;
    const bool phase1 = (wave < 8);
    const int gate  = phase1 ? (wave >> 2) : 2;
    const int mtile = wave & 1;
    const int ntile = (wave >> 1) & 1;
    const int n0 = nbase + ntile * 16;

    __shared__ float z_lds[GRU_B][33];
    __shared__ __align__(16) bf16_t hbuf[GRU_B][GRU_H + 8];  // reused for h and r*h

    // cooperative async copy [32][512] bf16 -> LDS (2048 x 16B chunks)
    auto copy_to_lds = [&](const bf16_t* __restrict__ src) {
        for (int c = threadIdx.x; c < GRU_B * (GRU_H / 8); c += SCAN_THREADS) {
            int row = c >> 6, kc = c & 63;
            async_b128_to_lds(&hbuf[row][kc * 8], src + row * GRU_H + kc * 8);
        }
        wait_asynccnt0();
        __syncthreads();
    };

    const bf16_t* W = (gate == 0) ? Wzh : (gate == 1) ? Wrh : Whh;
    v16bf wf[16];
#pragma unroll
    for (int kf = 0; kf < 16; ++kf)
        wf[kf] = load_frag(W, GRU_H, n0, kf * 32, lane);

    const int rlo = lane & 15, half = lane >> 4;
    unsigned seq = 0;

    for (int t = 0; t < GRU_S; ++t) {
        copy_to_lds(h_bf);                     // h(t-1), fresh after barrier #2
        if (phase1) {
            v8f acc = {};
#pragma unroll
            for (int kf = 0; kf < 16; ++kf) {
                v16bf a = load_frag(&hbuf[0][0], GRU_H + 8, mtile * 16, kf * 32, lane);
                acc = wmma_bf16(a, wf[kf], acc);
            }
#pragma unroll
            for (int rr = 0; rr < 8; ++rr) {
                int m  = mtile * 16 + rr + half * 8;   // batch row
                int nl = ntile * 16 + rlo;             // local column
                int n  = nbase + nl;
                float pre = gates[(((size_t)gate * GRU_B + m) * GRU_S + t) * GRU_H + n];
                float v = fast_sigmoid(pre + acc[rr]);
                if (gate == 0) {
                    z_lds[m][nl] = v;
                } else {
                    float rh = v * h_f32[m * GRU_H + n];
                    rh_bf[m * GRU_H + n] = to_bf16(rh);
                }
            }
        }
        grid_barrier(ctr, (++seq) * gridDim.x);   // r*h fully published

        copy_to_lds(rh_bf);
        if (!phase1) {
            v8f acc = {};
#pragma unroll
            for (int kf = 0; kf < 16; ++kf) {
                v16bf a = load_frag(&hbuf[0][0], GRU_H + 8, mtile * 16, kf * 32, lane);
                acc = wmma_bf16(a, wf[kf], acc);
            }
#pragma unroll
            for (int rr = 0; rr < 8; ++rr) {
                int m  = mtile * 16 + rr + half * 8;
                int nl = ntile * 16 + rlo;
                int n  = nbase + nl;
                float xh = gates[(((size_t)2 * GRU_B + m) * GRU_S + t) * GRU_H + n];
                float ht = tanhf(xh + acc[rr]);
                float z  = z_lds[m][nl];
                float ho = h_f32[m * GRU_H + n];
                float hn = z * ho + (1.0f - z) * ht;
                h_f32[m * GRU_H + n] = hn;
                h_bf[m * GRU_H + n]  = to_bf16(hn);
                size_t yi = ((size_t)m * GRU_S + t) * GRU_H + n;
                if (y_bf)  y_bf[yi]  = to_bf16(hn);
                if (y_f32) y_f32[yi] = hn;
                if (t == GRU_S - 1) hn_out[m * GRU_H + n] = hn;
            }
        }
        grid_barrier(ctr, (++seq) * gridDim.x);   // h fully updated
    }
}

// ---------------- host launch ----------------

extern "C" void kernel_launch(void* const* d_in, const int* in_sizes, int n_in,
                              void* d_out, int out_size, void* d_ws, size_t ws_size,
                              hipStream_t stream) {
    (void)in_sizes; (void)n_in; (void)out_size; (void)ws_size;
    const int B = GRU_B, S = GRU_S, IN = GRU_IN, H = GRU_H;
    const size_t MS = (size_t)B * S;

    const float* x   = (const float*)d_in[0];
    const float* hin = (const float*)d_in[1];
    const float* Wz0 = (const float*)d_in[2];  const float* bz0 = (const float*)d_in[3];
    const float* Wr0 = (const float*)d_in[4];  const float* br0 = (const float*)d_in[5];
    const float* Wh0 = (const float*)d_in[6];  const float* bh0 = (const float*)d_in[7];
    const float* Wz1 = (const float*)d_in[8];  const float* bz1 = (const float*)d_in[9];
    const float* Wr1 = (const float*)d_in[10]; const float* br1 = (const float*)d_in[11];
    const float* Wh1 = (const float*)d_in[12]; const float* bh1 = (const float*)d_in[13];
    float* out = (float*)d_out;

    char* ws = (char*)d_ws;
    size_t off = 0;
    auto alloc = [&](size_t bytes) -> char* {
        char* p = ws + off;
        off = (off + bytes + 255) & ~(size_t)255;
        return p;
    };

    bf16_t* xbf  = (bf16_t*)alloc(MS * IN * 2);
    bf16_t* Wzx0 = (bf16_t*)alloc((size_t)H * IN * 2);
    bf16_t* Wrx0 = (bf16_t*)alloc((size_t)H * IN * 2);
    bf16_t* Whx0 = (bf16_t*)alloc((size_t)H * IN * 2);
    bf16_t* Wzh0 = (bf16_t*)alloc((size_t)H * H * 2);
    bf16_t* Wrh0 = (bf16_t*)alloc((size_t)H * H * 2);
    bf16_t* Whh0 = (bf16_t*)alloc((size_t)H * H * 2);
    bf16_t* Wzx1 = (bf16_t*)alloc((size_t)H * H * 2);
    bf16_t* Wrx1 = (bf16_t*)alloc((size_t)H * H * 2);
    bf16_t* Whx1 = (bf16_t*)alloc((size_t)H * H * 2);
    bf16_t* Wzh1 = (bf16_t*)alloc((size_t)H * H * 2);
    bf16_t* Wrh1 = (bf16_t*)alloc((size_t)H * H * 2);
    bf16_t* Whh1 = (bf16_t*)alloc((size_t)H * H * 2);
    float*  gates = (float*)alloc((size_t)3 * MS * H * 4);   // reused by both layers
    bf16_t* y0    = (bf16_t*)alloc(MS * H * 2);
    bf16_t* hbf   = (bf16_t*)alloc((size_t)B * H * 2);
    float*  hf    = (float*)alloc((size_t)B * H * 4);
    bf16_t* rhbf  = (bf16_t*)alloc((size_t)B * H * 2);
    unsigned* ctr = (unsigned*)alloc(256);

    // ---- convert x and weights to bf16 (weights transposed to [n][k]) ----
    {
        int nx = (int)(MS * IN);
        f32_to_bf16_kernel<<<(nx + 255) / 256, 256, 0, stream>>>(x, xbf, nx);
    }
    const int g256 = (IN * H + 255) / 256;   // 256x512
    const int g512 = (H * H + 255) / 256;    // 512x512
    wconv_kernel<<<g256, 256, 0, stream>>>(Wz0,                 Wzx0, IN, H);
    wconv_kernel<<<g512, 256, 0, stream>>>(Wz0 + (size_t)IN * H, Wzh0, H, H);
    wconv_kernel<<<g256, 256, 0, stream>>>(Wr0,                 Wrx0, IN, H);
    wconv_kernel<<<g512, 256, 0, stream>>>(Wr0 + (size_t)IN * H, Wrh0, H, H);
    wconv_kernel<<<g256, 256, 0, stream>>>(Wh0,                 Whx0, IN, H);
    wconv_kernel<<<g512, 256, 0, stream>>>(Wh0 + (size_t)IN * H, Whh0, H, H);
    wconv_kernel<<<g512, 256, 0, stream>>>(Wz1,                 Wzx1, H, H);
    wconv_kernel<<<g512, 256, 0, stream>>>(Wz1 + (size_t)H * H,  Wzh1, H, H);
    wconv_kernel<<<g512, 256, 0, stream>>>(Wr1,                 Wrx1, H, H);
    wconv_kernel<<<g512, 256, 0, stream>>>(Wr1 + (size_t)H * H,  Wrh1, H, H);
    wconv_kernel<<<g512, 256, 0, stream>>>(Wh1,                 Whx1, H, H);
    wconv_kernel<<<g512, 256, 0, stream>>>(Wh1 + (size_t)H * H,  Whh1, H, H);

    dim3 gemm_grid((unsigned)(MS / 64), (unsigned)(H / 128));

    // ================= layer 0 =================
    hinit_kernel<<<(B * H + 255) / 256, 256, 0, stream>>>(hin, hbf, hf, ctr);
    gate_gemm_kernel<<<gemm_grid, 256, 0, stream>>>(xbf, Wzx0, bz0, gates + 0 * MS * H, (int)MS, IN);
    gate_gemm_kernel<<<gemm_grid, 256, 0, stream>>>(xbf, Wrx0, br0, gates + 1 * MS * H, (int)MS, IN);
    gate_gemm_kernel<<<gemm_grid, 256, 0, stream>>>(xbf, Whx0, bh0, gates + 2 * MS * H, (int)MS, IN);
    gru_scan_kernel<<<SCAN_BLOCKS, SCAN_THREADS, 0, stream>>>(
        gates, Wzh0, Wrh0, Whh0, hbf, hf, rhbf,
        /*y_bf=*/y0, /*y_f32=*/nullptr,
        /*hn_out=*/out + (size_t)B * S * H, ctr);

    // ================= layer 1 =================
    hinit_kernel<<<(B * H + 255) / 256, 256, 0, stream>>>(hin + (size_t)B * H, hbf, hf, ctr);
    gate_gemm_kernel<<<gemm_grid, 256, 0, stream>>>(y0, Wzx1, bz1, gates + 0 * MS * H, (int)MS, H);
    gate_gemm_kernel<<<gemm_grid, 256, 0, stream>>>(y0, Wrx1, br1, gates + 1 * MS * H, (int)MS, H);
    gate_gemm_kernel<<<gemm_grid, 256, 0, stream>>>(y0, Whx1, bh1, gates + 2 * MS * H, (int)MS, H);
    gru_scan_kernel<<<SCAN_BLOCKS, SCAN_THREADS, 0, stream>>>(
        gates, Wzh1, Wrh1, Whh1, hbf, hf, rhbf,
        /*y_bf=*/nullptr, /*y_f32=*/out,
        /*hn_out=*/out + (size_t)B * S * H + (size_t)B * H, ctr);
}
